// LNNLayer_12799002542740
// MI455X (gfx1250) — matrix operand order repaired and checked by reading, not compiled, not run on hardware
//
#include <hip/hip_runtime.h>
#include <math.h>

// ---------------------------------------------------------------------------
// ESN / liquid layer for MI455X (gfx1250, wave32, WMMA bf16->f32).
//   x:[32,1024,256] f32, W:[1024,1024] f32, W_in:[256,1024] f32 -> ys:[32,1024,1024] f32
//   h_{t+1} = 0.8*h_t + 0.2*tanh(x_t@W_in + h_t@W)
//
// Strategy:
//  * pack W, W_in into bf16 B-fragment-major layout; pack x into bf16
//    A-fragment-major layout (prep kernels, memory bound, ~20MB total).
//  * persistent scan kernel: 32 WGs x 128 thr = 128 waves = 2 m-groups x 64
//    n-tiles. Each wave keeps its 32KB W slice + 8KB W_in slice in LDS for the
//    entire run, keeps its 16x16 f32 h-slab in registers, and does
//    40 v_wmma_f32_16x16x32_bf16 per step over 4 independent acc chains.
//  * h state ping-pongs via a 128KB global (L2-hot) buffer; steps separated by
//    an atomic grid barrier with release/acquire fences.
// ---------------------------------------------------------------------------

typedef __attribute__((ext_vector_type(16))) __bf16 v16bf;
typedef __attribute__((ext_vector_type(8)))  float  v8f;

struct alignas(16) U4 { unsigned x, y, z, w; };
struct alignas(32) Q2 { U4 lo, hi; };

static constexpr int B = 32, T = 1024, D = 256, U = 1024;
static constexpr int NT = U / 16;        // 64 n-tiles
static constexpr int KT_W = U / 32;      // 32 k-tiles for h@W
static constexpr int KT_X = D / 32;      // 8  k-tiles for x@W_in
static constexpr int NWG = 32;           // persistent grid
static constexpr int WAVES_PER_WG = 4;   // 128 threads
static constexpr float LEAK = 0.2f;

// workspace layout (bytes)
static constexpr size_t OFF_BAR  = 0;
static constexpr size_t OFF_HB   = 256;                                   // 2 x [32][1024] bf16
static constexpr size_t OFF_WB   = OFF_HB  + (size_t)2 * B * U * 2;       // 131328
static constexpr size_t OFF_WINB = OFF_WB  + (size_t)NT * KT_W * 512 * 2; // +2MB
static constexpr size_t OFF_XB   = OFF_WINB+ (size_t)NT * KT_X * 512 * 2; // +512KB
static constexpr size_t WS_NEED  = OFF_XB  + (size_t)2 * T * KT_X * 512 * 2; // +16MB ~ 18.6MB

__device__ __forceinline__ v8f wmma_bf16(v16bf a, v16bf b, v8f c) {
  return __builtin_amdgcn_wmma_f32_16x16x32_bf16(false, a, false, b,
                                                 (short)0, c, false, false);
}

// two independent 16B chunks -> one 16-half fragment
__device__ __forceinline__ v16bf ld_frag2(const __bf16* p0, const __bf16* p1) {
  Q2 q;
  q.lo = *(const U4*)p0;
  q.hi = *(const U4*)p1;
  return __builtin_bit_cast(v16bf, q);
}

// ---------------- prep kernels ----------------

// W [1024x1024] row-major f32 -> B-fragment-major bf16:
// Wb[((nt*KT_W + kt)*32 + lane)*16 + i] = W[kt*32 + (lane>>4)*8 + (i&7) + (i>=8 ? 16 : 0)][nt*16 + (lane&15)]
__global__ void pack_W_kernel(const float* __restrict__ W, __bf16* __restrict__ Wb) {
  int idx = blockIdx.x * blockDim.x + threadIdx.x;     // 0 .. 1048576
  int i    = idx & 15;
  int lane = (idx >> 4) & 31;
  int kt   = (idx >> 9) & (KT_W - 1);
  int nt   = idx >> 14;
  int k = kt * 32 + ((lane >> 4) << 3) + (i & 7) + ((i >> 3) << 4);
  int n = nt * 16 + (lane & 15);
  Wb[idx] = (__bf16)W[(size_t)k * U + n];
}

// W_in [256x1024] row-major f32 -> B-fragment-major bf16 (8 k-tiles)
__global__ void pack_Win_kernel(const float* __restrict__ Win, __bf16* __restrict__ Winb) {
  int idx = blockIdx.x * blockDim.x + threadIdx.x;     // 0 .. 262144
  int i    = idx & 15;
  int lane = (idx >> 4) & 31;
  int kt   = (idx >> 9) & (KT_X - 1);
  int nt   = idx >> 12;
  int k = kt * 32 + ((lane >> 4) << 3) + (i & 7) + ((i >> 3) << 4);
  int n = nt * 16 + (lane & 15);
  Winb[idx] = (__bf16)Win[(size_t)k * U + n];
}

// x [32][1024][256] f32 -> A-fragment-major bf16:
// xb[(((mt*T + t)*KT_X + kt)*32 + lane)*16 + i] =
//   x[mt*16 + (lane&15)][t][kt*32 + (lane>>4)*8 + (i&7) + (i>=8 ? 16 : 0)]
__global__ void pack_x_kernel(const float* __restrict__ x, __bf16* __restrict__ xb) {
  int idx = blockIdx.x * blockDim.x + threadIdx.x;     // 0 .. 8388608
  int i    = idx & 15;
  int lane = (idx >> 4) & 31;
  int kt   = (idx >> 9) & (KT_X - 1);
  int t    = (idx >> 12) & (T - 1);
  int mt   = idx >> 22;
  int b = mt * 16 + (lane & 15);
  int d = kt * 32 + ((lane >> 4) << 3) + (i & 7) + ((i >> 3) << 4);
  xb[idx] = (__bf16)x[((size_t)b * T + t) * D + d];
}

__global__ void init_state_kernel(__bf16* __restrict__ hb, unsigned* __restrict__ bar) {
  int idx = blockIdx.x * blockDim.x + threadIdx.x;     // 0 .. 65536
  hb[idx] = (__bf16)0.0f;
  if (idx == 0) bar[0] = 0u;   // arrival count; bar[1] (generation) is monotonic
}

// ---------------- persistent scan kernel ----------------

__global__ __launch_bounds__(WAVES_PER_WG * 32) void esn_scan_kernel(
    const __bf16* __restrict__ xb,
    const __bf16* __restrict__ Wb,
    const __bf16* __restrict__ Winb,
    __bf16* __restrict__ hb,          // 2 x [32][1024] bf16 ping-pong
    float* __restrict__ ys,           // [32][1024][1024] f32
    unsigned* __restrict__ bar) {
  extern __shared__ __bf16 lds[];     // per wave: 16384 (W) + 4096 (Win) halves

  const int lane = threadIdx.x & 31;
  const int wave = threadIdx.x >> 5;
  const int gw = blockIdx.x * WAVES_PER_WG + wave;   // 0..127
  const int mt = gw >> 6;                            // 0..1 (m-group)
  const int nt = gw & 63;                            // 0..63 (n-tile)

  __bf16* sW   = lds + (size_t)wave * 20480;
  __bf16* sWin = sW + 16384;

  // stage this wave's W / W_in slices into LDS (held for the whole run)
  {
    const U4* src = (const U4*)(Wb + (size_t)nt * KT_W * 512);
    U4* dst = (U4*)sW;
    for (int i = lane; i < 2048; i += 32) dst[i] = src[i];
    const U4* src2 = (const U4*)(Winb + (size_t)nt * KT_X * 512);
    U4* dst2 = (U4*)sWin;
    for (int i = lane; i < 512; i += 32) dst2[i] = src2[i];
  }
  __syncthreads();

  const int mr = lane & 15;   // row within 16-row tile (A/B/C lane mapping)
  const int hi = lane >> 4;   // half-wave selector
  const size_t hrow = (size_t)(mt * 16 + mr) * U;    // h is [32][1024] bf16

  float hreg[8];
#pragma unroll
  for (int r = 0; r < 8; ++r) hreg[r] = 0.0f;

  for (int t = 0; t < T; ++t) {
    const __bf16* hsrc = hb + (size_t)(t & 1) * (B * U);

    v8f acc[4];
#pragma unroll
    for (int j = 0; j < 4; ++j) acc[j] = (v8f){0.f, 0.f, 0.f, 0.f, 0.f, 0.f, 0.f, 0.f};

    // h @ W  (K = 1024, 32 k-tiles, 4 independent accumulator chains)
#pragma unroll
    for (int kt = 0; kt < KT_W; ++kt) {
      const __bf16* pa = hsrc + hrow + kt * 32 + hi * 8;   // rows of h, bf16
      v16bf av = ld_frag2(pa, pa + 16);
      const __bf16* pb = sW + kt * 512 + lane * 16;        // LDS B-fragment
      v16bf bv = ld_frag2(pb, pb + 8);
      acc[kt & 3] = wmma_bf16(av, bv, acc[kt & 3]);
    }
    // x_t @ W_in  (K = 256, 8 k-tiles), fused input projection
    const __bf16* xfr = xb + ((size_t)mt * T + t) * (KT_X * 512);
#pragma unroll
    for (int kt = 0; kt < KT_X; ++kt) {
      const __bf16* pa = xfr + kt * 512 + lane * 16;       // packed A-fragment
      v16bf av = ld_frag2(pa, pa + 8);
      const __bf16* pb = sWin + kt * 512 + lane * 16;
      v16bf bv = ld_frag2(pb, pb + 8);
      acc[kt & 3] = wmma_bf16(av, bv, acc[kt & 3]);
    }

    // leaky tanh update (f32 state kept in registers), write ys + next-h
    __bf16* hdst = hb + (size_t)((t + 1) & 1) * (B * U);
#pragma unroll
    for (int r = 0; r < 8; ++r) {
      float u = acc[0][r] + acc[1][r] + acc[2][r] + acc[3][r];
      float hn = (1.0f - LEAK) * hreg[r] + LEAK * tanhf(u);
      hreg[r] = hn;
      int mrow = hi * 8 + r;
      size_t orow = ((size_t)(mt * 16 + mrow) * T + t) * U + nt * 16 + mr;
      __builtin_nontemporal_store(hn, ys + orow);          // 128MB stream, keep L2 clean
      hdst[(size_t)(mt * 16 + mrow) * U + nt * 16 + mr] = (__bf16)hn;
    }

    // ---- grid barrier: release stores, sync 32 WGs, acquire for next step ----
    __threadfence();         // make this thread's h stores agent-visible
    __syncthreads();         // all waves of this WG have released before leader arrives
    if (threadIdx.x == 0) {
      unsigned gen = __hip_atomic_load(&bar[1], __ATOMIC_RELAXED, __HIP_MEMORY_SCOPE_AGENT);
      unsigned arr = __hip_atomic_fetch_add(&bar[0], 1u, __ATOMIC_ACQ_REL, __HIP_MEMORY_SCOPE_AGENT);
      if (arr == NWG - 1) {
        __hip_atomic_store(&bar[0], 0u, __ATOMIC_RELAXED, __HIP_MEMORY_SCOPE_AGENT);
        __hip_atomic_fetch_add(&bar[1], 1u, __ATOMIC_RELEASE, __HIP_MEMORY_SCOPE_AGENT);
      } else {
        while (__hip_atomic_load(&bar[1], __ATOMIC_ACQUIRE, __HIP_MEMORY_SCOPE_AGENT) == gen) {
          __builtin_amdgcn_s_sleep(1);
        }
      }
    }
    __syncthreads();
    __threadfence();         // acquire: drop stale h lines from WGP$ before next reads
  }
}

// ---------------- host launcher ----------------

extern "C" void kernel_launch(void* const* d_in, const int* in_sizes, int n_in,
                              void* d_out, int out_size, void* d_ws, size_t ws_size,
                              hipStream_t stream) {
  (void)in_sizes; (void)n_in; (void)out_size;
  if (ws_size < WS_NEED) return;   // cannot run without scratch; avoid OOB

  const float* x   = (const float*)d_in[0];
  const float* W   = (const float*)d_in[1];
  const float* Win = (const float*)d_in[2];
  float* ys = (float*)d_out;

  char* ws = (char*)d_ws;
  unsigned* bar = (unsigned*)(ws + OFF_BAR);
  __bf16* hb    = (__bf16*)(ws + OFF_HB);
  __bf16* Wb    = (__bf16*)(ws + OFF_WB);
  __bf16* Winb  = (__bf16*)(ws + OFF_WINB);
  __bf16* xb    = (__bf16*)(ws + OFF_XB);

  pack_W_kernel  <<<4096, 256, 0, stream>>>(W, Wb);
  pack_Win_kernel<<<1024, 256, 0, stream>>>(Win, Winb);
  pack_x_kernel  <<<32768, 256, 0, stream>>>(x, xb);
  init_state_kernel<<<256, 256, 0, stream>>>(hb, bar);

  // 32 persistent WGs x 128 threads; 160KB dynamic LDS per WG (W + W_in slices)
  esn_scan_kernel<<<NWG, WAVES_PER_WG * 32, 20480 * WAVES_PER_WG * sizeof(__bf16), stream>>>(
      xb, Wb, Winb, hb, ys, bar);
}
